// BinaryClassifier_30090540875824
// MI455X (gfx1250) — compile-verified, matrix operand order checked by reading, not run on hardware
//
#include <hip/hip_runtime.h>
#include <hip/hip_fp16.h>
#include <math.h>
#include <stdint.h>

typedef __attribute__((ext_vector_type(16))) _Float16 v16h;
typedef __attribute__((ext_vector_type(8)))  _Float16 v8h;
typedef __attribute__((ext_vector_type(8)))  float    v8f;
typedef __attribute__((ext_vector_type(4)))  int      v4i;

#define FRAMES 31
#define NFFT   2048
#define BATCH  1024

// ---------- typed address-space pointers for the async builtin -------------
// Builtin signature (from probe diagnostic): param0 = int4 __device__* (AS1),
// param1 = int4 __shared__* (AS3). Build them via integer casts (always legal;
// on AMDGPU the LDS offset is the low 32 bits of a generic pointer).
typedef __attribute__((address_space(1))) v4i gm_v4i;
typedef __attribute__((address_space(3))) v4i lds_v4i;

__device__ __forceinline__ gm_v4i* to_gm128(const void* p) {
    return (gm_v4i*)(uintptr_t)p;
}
__device__ __forceinline__ lds_v4i* to_lds128(const void* p) {
    return (lds_v4i*)(uint32_t)(uintptr_t)p;
}

// ---------- CDNA5 async global->LDS copy (ASYNCcnt path), with fallback ----
__device__ __forceinline__ void async_copy_b128(const _Float16* g, _Float16* l) {
#if defined(__has_builtin) && __has_builtin(__builtin_amdgcn_global_load_async_to_lds_b128)
    __builtin_amdgcn_global_load_async_to_lds_b128(to_gm128(g), to_lds128(l), 0, 0);
#else
    *(v8h*)l = *(const v8h*)g;   // synchronous fallback
#endif
}
__device__ __forceinline__ void wait_async0() {
#if defined(__has_builtin) && __has_builtin(__builtin_amdgcn_s_wait_asynccnt)
    __builtin_amdgcn_s_wait_asynccnt(0);
#else
    asm volatile("s_wait_asynccnt 0x0" ::: "memory");
#endif
}

// ---------------- Stage A: frame mean (bandwidth bound: 260 MB read) -------
__global__ __launch_bounds__(256) void k_frame_mean(const float* __restrict__ x,
                                                    _Float16* __restrict__ xm) {
    int idx = blockIdx.x * 256 + threadIdx.x;     // 0 .. 1024*2048-1
    int b = idx >> 11;
    int n = idx & 2047;
    const float* p = x + (size_t)b * (FRAMES * NFFT) + n;
    float s = 0.f;
#pragma unroll
    for (int f = 0; f < FRAMES; ++f) s += p[(size_t)f * NFFT];
    xm[idx] = (_Float16)(s * (1.0f / FRAMES));
}

// ---------------- Stage B: cosine (real-DFT) matrix, symmetric -------------
__global__ __launch_bounds__(256) void k_cos_fill(_Float16* __restrict__ C) {
    int idx = blockIdx.x * 256 + threadIdx.x;     // 0 .. 2048*2048-1
    int n = idx >> 11;
    int k = idx & 2047;
    int m = (n * k) & 2047;
    float ang = (float)m * 3.0679615757712823e-3f;  // pi/1024
    C[idx] = (_Float16)__cosf(ang);
}

// ---------------- fp32 -> fp16 weight conversion ---------------------------
__global__ __launch_bounds__(256) void k_f32_to_f16(const float* __restrict__ src,
                                                    _Float16* __restrict__ dst, int n) {
    int i = blockIdx.x * 256 + threadIdx.x;
    if (i < n) dst[i] = (_Float16)src[i];
}

// ---------------- WMMA GEMM: D = act(A[MxK] @ Wt[NxK]^T + bias) ------------
// 256 threads = 8 wave32; block tile 128x128, wave tile 32(M)x64(N).
// Double-buffered LDS tiles filled by GLOBAL_LOAD_ASYNC_TO_LDS_B128
// (ASYNCcnt), consumed via DS_LOAD_B128 fragment reads. LDS row stride of
// 40 halves (80 B) keeps b128 reads 16B-aligned and bank-conflict-free.
#define LDSK 40
template <bool RELU, bool BIAS>
__global__ __launch_bounds__(256) void k_gemm_f16(const _Float16* __restrict__ A,
                                                  const _Float16* __restrict__ Wt,
                                                  const float* __restrict__ bias,
                                                  _Float16* __restrict__ D,
                                                  int M, int N, int K) {
    __shared__ _Float16 lA[2][128][LDSK];   // 20 KB
    __shared__ _Float16 lB[2][128][LDSK];   // 20 KB

    const int t    = threadIdx.x;
    const int lane = t & 31;
    const int wid  = t >> 5;
    const int wm = wid & 3;          // 4 waves stacked in M
    const int wn = wid >> 2;         // 2 waves in N
    const int m0w = wm * 32;         // LDS-local wave M offset
    const int n0w = wn * 64;         // LDS-local wave N offset

    const _Float16* Ga = A  + (size_t)(blockIdx.y * 128) * K;   // 128 A rows
    const _Float16* Gb = Wt + (size_t)(blockIdx.x * 128) * K;   // 128 Wt rows

    // staging assignment: 128 rows x 4 b128-segments = 512 chunks per tile;
    // each of 256 threads copies 2 chunks of A and 2 of B per K-step.
    const int r0 = t >> 2;           // 0..63
    const int sg = t & 3;            // segment (16B) within 32-half row slice

    const int r    = lane & 15;
    const int koff = (lane >> 4) << 3;   // 0 or 8 (f16 A/B fragment K layout)

    v8f acc[2][4] = {};

    auto stage = [&](int buf, int k0) {
        const _Float16* ga = Ga + (size_t)r0 * K + k0 + sg * 8;
        async_copy_b128(ga,                  &lA[buf][r0][sg * 8]);
        async_copy_b128(ga + (size_t)64 * K, &lA[buf][r0 + 64][sg * 8]);
        const _Float16* gb = Gb + (size_t)r0 * K + k0 + sg * 8;
        async_copy_b128(gb,                  &lB[buf][r0][sg * 8]);
        async_copy_b128(gb + (size_t)64 * K, &lB[buf][r0 + 64][sg * 8]);
    };

    stage(0, 0);
    wait_async0();
    __syncthreads();

    int buf = 0;
    for (int k0 = 0; k0 < K; k0 += 32) {
        if (k0 + 32 < K) stage(buf ^ 1, k0 + 32);

        union Frag { v16h v; v8h h[2]; } a[2], b[4];
#pragma unroll
        for (int mi = 0; mi < 2; ++mi) {
            const _Float16* p = &lA[buf][m0w + mi * 16 + r][koff];
            a[mi].h[0] = *(const v8h*)(p);
            a[mi].h[1] = *(const v8h*)(p + 16);
        }
#pragma unroll
        for (int ni = 0; ni < 4; ++ni) {
            const _Float16* p = &lB[buf][n0w + ni * 16 + r][koff];
            b[ni].h[0] = *(const v8h*)(p);
            b[ni].h[1] = *(const v8h*)(p + 16);
        }
#pragma unroll
        for (int mi = 0; mi < 2; ++mi)
#pragma unroll
            for (int ni = 0; ni < 4; ++ni)
                acc[mi][ni] = __builtin_amdgcn_wmma_f32_16x16x32_f16(
                    false, a[mi].v, false, b[ni].v, (short)0, acc[mi][ni],
                    false, false);

        wait_async0();      // next buffer staged
        __syncthreads();    // all waves done reading current buffer
        buf ^= 1;
    }

    // Epilogue: C/D layout — lane holds column n=lane&15; rows (lane>>4)*8+rr.
    const int n0 = blockIdx.x * 128 + n0w;
    const int m0 = blockIdx.y * 128 + m0w;
    const int nlane = lane & 15;
    const int mbase = (lane >> 4) << 3;
#pragma unroll
    for (int ni = 0; ni < 4; ++ni) {
        const int col = n0 + ni * 16 + nlane;
        const float bv = BIAS ? bias[col] : 0.0f;
#pragma unroll
        for (int mi = 0; mi < 2; ++mi) {
#pragma unroll
            for (int rr = 0; rr < 8; ++rr) {
                const int row = m0 + mi * 16 + mbase + rr;
                float v = acc[mi][ni][rr] + bv;
                if (RELU) v = v > 0.0f ? v : 0.0f;
                D[(size_t)row * N + col] = (_Float16)v;
            }
        }
    }
}

// ---------------- final: out[b] = sigmoid(dot(h2[b,:], W3) + b3) -----------
__global__ __launch_bounds__(256) void k_final(const _Float16* __restrict__ h2,
                                               const float* __restrict__ W3,
                                               const float* __restrict__ b3,
                                               float* __restrict__ out) {
    __shared__ float red[256];
    const int b = blockIdx.x;
    const int t = threadIdx.x;
    red[t] = (float)h2[(size_t)b * 256 + t] * W3[t];
    __syncthreads();
    for (int s = 128; s > 0; s >>= 1) {
        if (t < s) red[t] += red[t + s];
        __syncthreads();
    }
    if (t == 0) out[b] = 1.0f / (1.0f + expf(-(red[0] + b3[0])));
}

extern "C" void kernel_launch(void* const* d_in, const int* in_sizes, int n_in,
                              void* d_out, int out_size, void* d_ws, size_t ws_size,
                              hipStream_t stream) {
    (void)in_sizes; (void)n_in; (void)out_size; (void)ws_size;

    const float* x  = (const float*)d_in[0];
    const float* W1 = (const float*)d_in[1];
    const float* b1 = (const float*)d_in[2];
    const float* W2 = (const float*)d_in[3];
    const float* b2 = (const float*)d_in[4];
    const float* W3 = (const float*)d_in[5];
    const float* b3 = (const float*)d_in[6];
    float* out = (float*)d_out;

    char* ws = (char*)d_ws;                    // ~18.2 MiB used, all 16B-aligned
    _Float16* xm  = (_Float16*)(ws);                                  // 4 MiB
    _Float16* C   = (_Float16*)(ws + ( 4u << 20));                    // 8 MiB
    _Float16* H   = (_Float16*)(ws + (12u << 20));                    // 4 MiB
    _Float16* W1h = (_Float16*)(ws + (16u << 20));                    // 1 MiB
    _Float16* W2h = (_Float16*)(ws + (17u << 20));                    // 128 KiB
    _Float16* h1  = (_Float16*)(ws + (17u << 20) + (256u << 10));     // 512 KiB
    _Float16* h2  = (_Float16*)(ws + (17u << 20) + (768u << 10));     // 512 KiB

    // Stage A: frame mean (linearity: mean of FFTs == FFT of mean)
    k_frame_mean<<<(BATCH * NFFT) / 256, 256, 0, stream>>>(x, xm);
    // Stage B: cosine matrix (Re(FFT) == cosine transform)
    k_cos_fill<<<(NFFT * NFFT) / 256, 256, 0, stream>>>(C);
    // Weight conversions
    k_f32_to_f16<<<(256 * 2048) / 256, 256, 0, stream>>>(W1, W1h, 256 * 2048);
    k_f32_to_f16<<<(256 * 256) / 256, 256, 0, stream>>>(W2, W2h, 256 * 256);

    // H = xm @ C          (C symmetric -> serves as [N][K] operand)
    k_gemm_f16<false, false><<<dim3(2048 / 128, BATCH / 128), 256, 0, stream>>>(
        xm, C, nullptr, H, BATCH, 2048, 2048);
    // h1 = relu(H @ W1^T + b1)
    k_gemm_f16<true, true><<<dim3(256 / 128, BATCH / 128), 256, 0, stream>>>(
        H, W1h, b1, h1, BATCH, 256, 2048);
    // h2 = relu(h1 @ W2^T + b2)
    k_gemm_f16<true, true><<<dim3(256 / 128, BATCH / 128), 256, 0, stream>>>(
        h1, W2h, b2, h2, BATCH, 256, 256);
    // out = sigmoid(h2 @ W3 + b3)
    k_final<<<BATCH, 256, 0, stream>>>(h2, W3, b3, out);
}
